// UNETR_MoE_CLIP_RCNN_27487790694798
// MI455X (gfx1250) — matrix operand order repaired and checked by reading, not compiled, not run on hardware
//
#include <hip/hip_runtime.h>
#include <hip/hip_bf16.h>

// ----------------------------- constants -------------------------------
#define BB   8
#define NTOK 576
#define DD   768
#define NH   12
#define DH   64
#define LL   12
#define EE   4
#define HF   3072
#define MTOT (BB * NTOK)      // 4608 rows
#define QKVD (3 * DD)         // 2304

typedef __attribute__((ext_vector_type(16))) _Float16 v16h;
typedef __attribute__((ext_vector_type(8)))  float    v8f;

enum {
  MODE_NONE     = 0,
  MODE_GATHER   = 1,   // A rows gathered via rowidx, limited by *cntPtr
  MODE_SCATTER  = 2,   // out rows scattered via rowidx, scaled by scale[], +=
  MODE_GELU     = 4,
  MODE_RESID    = 8,   // out = resid + acc
  MODE_POS      = 16,  // acc += pos[(row % NTOK)*Nc + col]
  MODE_A_F16    = 32,  // A matrix is _Float16 (raw async copy to LDS)
  MODE_OUT_F16  = 64,  // Out written as _Float16
};

// ------------------- async global->LDS copy (inline asm) ----------------
// ISA 15.18.3: GLOBAL_LOAD_ASYNC_TO_LDS_B128 — VDST holds per-lane LDS byte
// offset; VADDR is the 64-bit global address; tracked by ASYNCcnt.
#define HAVE_ASYNC_LDS 1
__device__ inline void async_copy_b128(const void* g, void* l) {
  unsigned int loff = (unsigned int)(uintptr_t)l;   // LDS aperture: low 32 bits
  asm volatile("global_load_async_to_lds_b128 %0, %1, off"
               :
               : "v"(loff), "v"(g)
               : "memory");
}
__device__ inline void wait_async_lds() {
  asm volatile("s_wait_asynccnt 0" ::: "memory");
}

// ----------------------- WMMA fragment helpers -------------------------
__device__ inline v8f wmma16(v16h a, v16h b, v8f c) {
  return __builtin_amdgcn_wmma_f32_16x16x32_f16(false, a, false, b, (short)0, c,
                                                false, false);
}

// A fragment 16x32 f16. base points at fragment row 0; ld in halves.
// Lane L: row = L%16; lanes 0-15 take K=0..15; lanes 16-31 K=16..31.
__device__ inline v16h load_frag_a(const _Float16* base, int ld, int lane, int koff) {
  int m  = lane & 15;
  int hi = lane >> 4;
  const _Float16* src = base + m * ld + koff + hi * 16;
  v16h a;
#pragma unroll
  for (int j = 0; j < 8; ++j) {
    int k0 = 2 * j;
    a[2 * j]     = src[k0];
    a[2 * j + 1] = src[k0 + 1];
  }
  return a;
}

// B fragment 32x16 f16 from a [n][k] tile (weight rows = output cols).
// Lane L: col n = L%16; lanes 0-15 K=0..15, lanes 16-31 K=16..31.
__device__ inline v16h load_frag_b(const _Float16* base, int ld, int lane) {
  int n     = lane & 15;
  int kbase = (lane >> 4) * 16;
  const _Float16* src = base + n * ld + kbase;
  v16h b;
#pragma unroll
  for (int j = 0; j < 8; ++j) {
    b[2 * j]     = src[2 * j];
    b[2 * j + 1] = src[2 * j + 1];
  }
  return b;
}

// --------------------------- generic GEMM ------------------------------
// Out[M,Nc] = epilogue( A[M,K] @ W[Nc,K]^T + bias )
#define BLKT 64
#define KSTEP 32
#define LDT 40   // padded LDS row stride in halves (80B: 16B-aligned rows)

__global__ __launch_bounds__(256)
void gemm_wmma_kernel(const float* __restrict__ A, const float* __restrict__ W,
                      const float* __restrict__ bias, const float* __restrict__ pos,
                      const float* __restrict__ resid, float* __restrict__ Out,
                      int M, int Nc, int K, int mode,
                      const int* __restrict__ rowidx, const int* __restrict__ cntPtr,
                      const float* __restrict__ scale) {
  __shared__ __align__(16) _Float16 As[BLKT * LDT];
  __shared__ __align__(16) _Float16 Ws[BLKT * LDT];

  int effM = M;
  if (cntPtr) { int c = *cntPtr; effM = c < M ? c : M; }
  int m0 = blockIdx.x * BLKT;
  int n0 = blockIdx.y * BLKT;
  if (m0 >= effM) return;

  int tid  = threadIdx.x;
  int wave = tid >> 5;
  int lane = tid & 31;
  int fr     = wave & 3;            // fragment row 0..3
  int fcbase = (wave >> 2) * 2;     // fragment cols {0,1} or {2,3}

  int lrow = tid >> 2;              // 0..63 tile row loaded by this thread
  int lcol = (tid & 3) * 8;         // 8 consecutive k per thread

  int agr  = m0 + lrow;
  int arow = -1;
  if (agr < effM) arow = (mode & MODE_GATHER) ? rowidx[agr] : agr;
  const float*    aptrF = nullptr;
  const _Float16* aptrH = nullptr;
  if (arow >= 0) {
    if (mode & MODE_A_F16) aptrH = (const _Float16*)A + (size_t)arow * K + lcol;
    else                   aptrF = A + (size_t)arow * K + lcol;
  }
  const float* wptr = W + (size_t)(n0 + lrow) * K + lcol;

  v8f acc0 = {};
  v8f acc1 = {};

  for (int kb = 0; kb < K; kb += KSTEP) {
    if (kb + KSTEP < K) {
      if (aptrF) __builtin_prefetch(aptrF + kb + KSTEP, 0, 0);
      if (aptrH) __builtin_prefetch(aptrH + kb + KSTEP, 0, 0);
      __builtin_prefetch(wptr + kb + KSTEP, 0, 0);
    }
    _Float16* as = As + lrow * LDT + lcol;
    if (mode & MODE_A_F16) {
      if (aptrH) {
        async_copy_b128(aptrH + kb, as);           // raw 8-half tile copy
      } else {
#pragma unroll
        for (int j = 0; j < 8; ++j) as[j] = (_Float16)0.f;
      }
    } else if (aptrF) {
      float4 v0 = *(const float4*)(aptrF + kb);
      float4 v1 = *(const float4*)(aptrF + kb + 4);
      as[0] = (_Float16)v0.x; as[1] = (_Float16)v0.y;
      as[2] = (_Float16)v0.z; as[3] = (_Float16)v0.w;
      as[4] = (_Float16)v1.x; as[5] = (_Float16)v1.y;
      as[6] = (_Float16)v1.z; as[7] = (_Float16)v1.w;
    } else {
#pragma unroll
      for (int j = 0; j < 8; ++j) as[j] = (_Float16)0.f;
    }
    _Float16* ws = Ws + lrow * LDT + lcol;
    {
      float4 v0 = *(const float4*)(wptr + kb);
      float4 v1 = *(const float4*)(wptr + kb + 4);
      ws[0] = (_Float16)v0.x; ws[1] = (_Float16)v0.y;
      ws[2] = (_Float16)v0.z; ws[3] = (_Float16)v0.w;
      ws[4] = (_Float16)v1.x; ws[5] = (_Float16)v1.y;
      ws[6] = (_Float16)v1.z; ws[7] = (_Float16)v1.w;
    }
    if (mode & MODE_A_F16) wait_async_lds();
    __syncthreads();
    v16h a  = load_frag_a(As + fr * 16 * LDT, LDT, lane, 0);
    v16h b0 = load_frag_b(Ws + fcbase * 16 * LDT, LDT, lane);
    v16h b1 = load_frag_b(Ws + (fcbase + 1) * 16 * LDT, LDT, lane);
    acc0 = wmma16(a, b0, acc0);
    acc1 = wmma16(a, b1, acc1);
    __syncthreads();
  }

  // epilogue
  int hi = lane >> 4;
  int cn = lane & 15;
#pragma unroll
  for (int pair = 0; pair < 2; ++pair) {
    v8f acc = pair ? acc1 : acc0;
    int col = n0 + (fcbase + pair) * 16 + cn;
    float bv = bias ? bias[col] : 0.f;
#pragma unroll
    for (int j = 0; j < 8; ++j) {
      int gr = m0 + fr * 16 + hi * 8 + j;
      if (gr >= effM) continue;
      float val = acc[j] + bv;
      if (mode & MODE_GELU) val = 0.5f * val * (1.f + erff(val * 0.70710678f));
      if (mode & MODE_POS)  val += pos[(size_t)(gr % NTOK) * Nc + col];
      if (mode & MODE_SCATTER) {
        int tr = rowidx[gr];
        Out[(size_t)tr * Nc + col] += scale[tr] * val;
      } else if (mode & MODE_RESID) {
        Out[(size_t)gr * Nc + col] = resid[(size_t)gr * Nc + col] + val;
      } else if (mode & MODE_OUT_F16) {
        ((_Float16*)Out)[(size_t)gr * Nc + col] = (_Float16)val;
      } else {
        Out[(size_t)gr * Nc + col] = val;
      }
    }
  }
}

// ---------------------------- attention --------------------------------
// One wave per (16 query rows, head, batch). WMMA for QK^T and P*V.
__global__ __launch_bounds__(32)
void attn_kernel(const float* __restrict__ qkv, float* __restrict__ out) {
  __shared__ __align__(16) _Float16 Qs[16 * 64];
  __shared__ float    Ss[16 * 576];
  __shared__ __align__(16) _Float16 Ps[16 * 592];

  int qt = blockIdx.x;   // 0..35
  int hh = blockIdx.y;   // 0..11
  int bb = blockIdx.z;   // 0..7
  int lane = threadIdx.x;

  const size_t rs = QKVD;
  const float* base = qkv + (size_t)bb * NTOK * rs;

  for (int i = lane; i < 16 * 64; i += 32) {
    int r = i >> 6, c = i & 63;
    Qs[i] = (_Float16)base[(size_t)(qt * 16 + r) * rs + hh * 64 + c];
  }
  __syncthreads();

  // scores = Q @ K^T * 1/8
  for (int kt = 0; kt < 36; ++kt) {
    v8f acc = {};
#pragma unroll
    for (int ks = 0; ks < 2; ++ks) {
      v16h a = load_frag_a(Qs, 64, lane, ks * 32);
      v16h b;
      int key = kt * 16 + (lane & 15);
      int kb0 = (lane >> 4) * 16 + ks * 32;
      const float* kp = base + (size_t)key * rs + DD + hh * 64 + kb0;
#pragma unroll
      for (int j = 0; j < 8; ++j) {
        b[2 * j]     = (_Float16)kp[2 * j];
        b[2 * j + 1] = (_Float16)kp[2 * j + 1];
      }
      acc = wmma16(a, b, acc);
    }
    int cn = kt * 16 + (lane & 15);
    int hi = lane >> 4;
#pragma unroll
    for (int j = 0; j < 8; ++j) Ss[(hi * 8 + j) * 576 + cn] = acc[j] * 0.125f;
  }
  __syncthreads();

  // row softmax
  for (int r = 0; r < 16; ++r) {
    float mx = -3.4e38f;
    for (int c = lane; c < 576; c += 32) mx = fmaxf(mx, Ss[r * 576 + c]);
    for (int o = 16; o; o >>= 1) mx = fmaxf(mx, __shfl_xor(mx, o));
    float sum = 0.f;
    for (int c = lane; c < 576; c += 32) {
      float e = __expf(Ss[r * 576 + c] - mx);
      Ss[r * 576 + c] = e;
      sum += e;
    }
    for (int o = 16; o; o >>= 1) sum += __shfl_xor(sum, o);
    float inv = 1.f / sum;
    for (int c = lane; c < 576; c += 32)
      Ps[r * 592 + c] = (_Float16)(Ss[r * 576 + c] * inv);
  }
  __syncthreads();

  // out = P @ V
  v8f oacc[4] = {{}, {}, {}, {}};
  for (int kk = 0; kk < 18; ++kk) {
    v16h a = load_frag_a(Ps, 592, lane, kk * 32);
#pragma unroll
    for (int fc = 0; fc < 4; ++fc) {
      v16h b;
      int dim = fc * 16 + (lane & 15);
      int tb  = kk * 32 + (lane >> 4) * 16;
      const float* vp = base + 2 * DD + hh * 64 + dim;
#pragma unroll
      for (int j = 0; j < 8; ++j) {
        b[2 * j]     = (_Float16)vp[(size_t)(tb + 2 * j) * rs];
        b[2 * j + 1] = (_Float16)vp[(size_t)(tb + 2 * j + 1) * rs];
      }
      oacc[fc] = wmma16(a, b, oacc[fc]);
    }
  }
  int hi = lane >> 4, cn = lane & 15;
#pragma unroll
  for (int fc = 0; fc < 4; ++fc) {
#pragma unroll
    for (int j = 0; j < 8; ++j) {
      int tok = qt * 16 + hi * 8 + j;
      out[((size_t)bb * NTOK + tok) * DD + hh * 64 + fc * 16 + cn] = oacc[fc][j];
    }
  }
}

// ---------------------------- layernorm --------------------------------
__global__ __launch_bounds__(256)
void ln_kernel(const float* __restrict__ x, const float* __restrict__ g,
               const float* __restrict__ b, float* __restrict__ out) {
  int tok  = blockIdx.x * 8 + (threadIdx.x >> 5);
  int lane = threadIdx.x & 31;
  if (tok >= MTOT) return;
  const float* row = x + (size_t)tok * DD;
  float s = 0.f, s2 = 0.f;
  for (int i = lane; i < DD; i += 32) {
    float v = row[i];
    s += v; s2 += v * v;
  }
  for (int o = 16; o; o >>= 1) { s += __shfl_xor(s, o); s2 += __shfl_xor(s2, o); }
  float m   = s * (1.f / DD);
  float var = s2 * (1.f / DD) - m * m;
  float inv = rsqrtf(var + 1e-5f);
  for (int i = lane; i < DD; i += 32)
    out[(size_t)tok * DD + i] = (row[i] - m) * inv * g[i] + b[i];
}

// ------------------------- gate + routing ------------------------------
__global__ void gate_kernel(const float* __restrict__ h2, const float* __restrict__ gw,
                            const float* __restrict__ gb, int* __restrict__ idx,
                            float* __restrict__ wgt) {
  int tok = blockIdx.x * 256 + threadIdx.x;
  if (tok >= MTOT) return;
  const float* row = h2 + (size_t)tok * DD;
  float best = -3.4e38f;
  int bi = 0;
  for (int e = 0; e < EE; ++e) {
    float s = gb[e];
    const float* w = gw + e * DD;
    for (int i = 0; i < DD; ++i) s += row[i] * w[i];
    if (s > best) { best = s; bi = e; }
  }
  idx[tok] = bi;
  wgt[tok] = best;
}

__global__ void zero_cnt_kernel(int* cnt) {
  if (threadIdx.x < EE) cnt[threadIdx.x] = 0;
}

__global__ void route_kernel(const int* __restrict__ idx, int* __restrict__ cnt,
                             int* __restrict__ bucket) {
  int tok = blockIdx.x * 256 + threadIdx.x;
  if (tok >= MTOT) return;
  int e   = idx[tok];
  int pos = atomicAdd(&cnt[e], 1);
  bucket[e * MTOT + pos] = tok;
}

// -------------------------- data movement ------------------------------
__global__ void im2col_kernel(const float* __restrict__ x, float* __restrict__ p) {
  int i = blockIdx.x * 256 + threadIdx.x;
  if (i >= MTOT * 256) return;
  int pp = i & 255;
  int n  = (i >> 8) % NTOK;
  int b  = i / (NTOK * 256);
  int hp = n / 24, wp = n % 24;
  int py = pp >> 4, px = pp & 15;
  p[i] = x[((size_t)b * 384 + hp * 16 + py) * 384 + wp * 16 + px];
}

__global__ void rearrange_kernel(const float* __restrict__ t, float* __restrict__ out) {
  int i = blockIdx.x * 256 + threadIdx.x;
  if (i >= BB * DD * NTOK) return;
  int wp = i % 24;
  int hp = (i / 24) % 24;
  int d  = (i / NTOK) % DD;
  int b  = i / (NTOK * DD);
  out[i] = t[((size_t)b * NTOK + hp * 24 + wp) * DD + d];
}

// ------------------------------ host -----------------------------------
extern "C" void kernel_launch(void* const* d_in, const int* in_sizes, int n_in,
                              void* d_out, int out_size, void* d_ws, size_t ws_size,
                              hipStream_t stream) {
  (void)in_sizes; (void)n_in; (void)out_size; (void)ws_size;
  const float* x        = (const float*)d_in[0];
  const float* patch_w  = (const float*)d_in[1];
  const float* patch_b  = (const float*)d_in[2];
  const float* pos_emb  = (const float*)d_in[3];
  const float* ln1_g    = (const float*)d_in[4];
  const float* ln1_b    = (const float*)d_in[5];
  const float* qkv_w    = (const float*)d_in[6];
  const float* qkv_b    = (const float*)d_in[7];
  const float* out_w    = (const float*)d_in[8];
  const float* out_b    = (const float*)d_in[9];
  const float* ln2_g    = (const float*)d_in[10];
  const float* ln2_b    = (const float*)d_in[11];
  const float* gate_w   = (const float*)d_in[12];
  const float* gate_b   = (const float*)d_in[13];
  const float* w1       = (const float*)d_in[14];
  const float* b1       = (const float*)d_in[15];
  const float* w2       = (const float*)d_in[16];
  const float* b2       = (const float*)d_in[17];
  float* out = (float*)d_out;

  float* wsf = (float*)d_ws;
  size_t off = 0;
  auto alloc = [&](size_t n) { float* p = wsf + off; off += (n + 63) & ~(size_t)63; return p; };
  float* t       = alloc((size_t)MTOT * DD);
  float* h       = alloc((size_t)MTOT * DD);
  float* qkvbuf  = alloc((size_t)MTOT * QKVD);
  float* attn    = alloc((size_t)MTOT * DD);
  float* h2      = alloc((size_t)MTOT * DD);
  float* hid     = alloc((size_t)MTOT * HF / 2 + 64);   // stored as _Float16
  float* patches = alloc((size_t)MTOT * 256);
  float* wgt     = alloc(MTOT);
  int*   idx     = (int*)alloc(MTOT);
  int*   cnt     = (int*)alloc(64);
  int*   bucket  = (int*)alloc((size_t)EE * MTOT);

  // patch embed
  im2col_kernel<<<(MTOT * 256 + 255) / 256, 256, 0, stream>>>(x, patches);
  {
    dim3 g(MTOT / BLKT, DD / BLKT);
    gemm_wmma_kernel<<<g, 256, 0, stream>>>(patches, patch_w, patch_b, pos_emb,
                                            nullptr, t, MTOT, DD, 256, MODE_POS,
                                            nullptr, nullptr, nullptr);
  }

  for (int l = 0; l < LL; ++l) {
    const float* l1g = ln1_g + l * DD;
    const float* l1b = ln1_b + l * DD;
    const float* qw  = qkv_w + (size_t)l * QKVD * DD;
    const float* qb  = qkv_b + (size_t)l * QKVD;
    const float* ow  = out_w + (size_t)l * DD * DD;
    const float* ob  = out_b + (size_t)l * DD;
    const float* l2g = ln2_g + l * DD;
    const float* l2b = ln2_b + l * DD;
    const float* gw  = gate_w + (size_t)l * EE * DD;
    const float* gb  = gate_b + (size_t)l * EE;

    ln_kernel<<<MTOT / 8, 256, 0, stream>>>(t, l1g, l1b, h);
    {
      dim3 g(MTOT / BLKT, QKVD / BLKT);
      gemm_wmma_kernel<<<g, 256, 0, stream>>>(h, qw, qb, nullptr, nullptr, qkvbuf,
                                              MTOT, QKVD, DD, MODE_NONE,
                                              nullptr, nullptr, nullptr);
    }
    {
      dim3 g(NTOK / 16, NH, BB);
      attn_kernel<<<g, 32, 0, stream>>>(qkvbuf, attn);
    }
    {
      dim3 g(MTOT / BLKT, DD / BLKT);
      gemm_wmma_kernel<<<g, 256, 0, stream>>>(attn, ow, ob, nullptr, t, t,
                                              MTOT, DD, DD, MODE_RESID,
                                              nullptr, nullptr, nullptr);
    }
    ln_kernel<<<MTOT / 8, 256, 0, stream>>>(t, l2g, l2b, h2);
    gate_kernel<<<(MTOT + 255) / 256, 256, 0, stream>>>(h2, gw, gb, idx, wgt);
    zero_cnt_kernel<<<1, 32, 0, stream>>>(cnt);
    route_kernel<<<(MTOT + 255) / 256, 256, 0, stream>>>(idx, cnt, bucket);

    for (int e = 0; e < EE; ++e) {
      const float* W1e = w1 + ((size_t)l * EE + e) * (size_t)HF * DD;
      const float* B1e = b1 + ((size_t)l * EE + e) * HF;
      const float* W2e = w2 + ((size_t)l * EE + e) * (size_t)DD * HF;
      const float* B2e = b2 + ((size_t)l * EE + e) * DD;
      const int* bkt = bucket + e * MTOT;
      const int* ce  = cnt + e;
      {
        // hid (f16) = gelu(h2[bucket] @ W1^T + b1)
        dim3 g(MTOT / BLKT, HF / BLKT);
        gemm_wmma_kernel<<<g, 256, 0, stream>>>(h2, W1e, B1e, nullptr, nullptr,
                                                hid, MTOT, HF, DD,
                                                MODE_GATHER | MODE_GELU | MODE_OUT_F16,
                                                bkt, ce, nullptr);
      }
      {
        // t[bucket] += wgt * (hid @ W2^T + b2); A is f16 -> async LDS copy
        dim3 g(MTOT / BLKT, DD / BLKT);
        gemm_wmma_kernel<<<g, 256, 0, stream>>>(hid, W2e, B2e, nullptr, nullptr,
                                                t, MTOT, DD, HF,
                                                MODE_SCATTER | MODE_A_F16,
                                                bkt, ce, wgt);
      }
    }
  }

  rearrange_kernel<<<(BB * DD * NTOK + 255) / 256, 256, 0, stream>>>(t, out);
}
